// BGNNContext_50525995270246
// MI455X (gfx1250) — compile-verified, matrix operand order checked by reading, not run on hardware
//
#include <hip/hip_runtime.h>
#include <hip/hip_bf16.h>

typedef __attribute__((ext_vector_type(16))) __bf16 v16bf;
typedef __attribute__((ext_vector_type(8)))  float  v8f;

#define NN 4096
#define EE 32768
#define PP 4096
#define HH 1024
#define FF 64

#define GF_RELU_IN  1
#define GF_RELU_OUT 2
#define GF_ACC      4

// ---------------------------------------------------------------------------
// transpose + fp32->bf16 convert:  src[K][Nv] f32  ->  dst[Nv][K] bf16
// ---------------------------------------------------------------------------
__global__ void transpose_cvt_kernel(const float* __restrict__ src,
                                     __bf16* __restrict__ dst,
                                     int K, int Nv) {
  long long i = (long long)blockIdx.x * 256 + threadIdx.x;
  long long tot = (long long)K * Nv;
  if (i >= tot) return;
  int k = (int)(i % K);
  int n = (int)(i / K);
  dst[(size_t)n * K + k] = (__bf16)src[(size_t)k * Nv + n];
}

__global__ void zero_f32_kernel(float* __restrict__ p, long long n) {
  long long i = (long long)blockIdx.x * 256 + threadIdx.x;
  if (i < n) p[i] = 0.f;
}

// ---------------------------------------------------------------------------
// Generic bf16-WMMA GEMM:  C[M,Nout] (fp32) = op(A[M,K] fp32) @ Wt^T + bias
//   Wt is the PRE-TRANSPOSED weight, bf16, [Nout][K] row-major.
//   FLAGS (compile-time): 1 relu(A), 2 relu(out), 4 C += result
// Block: 256 threads = 8 waves. Tile BM=128 (16 rows/wave), BN=64, BK=64.
// Double-buffered LDS: global loads for tile i+1 issued before computing
// tile i; one barrier per K-block.
// ---------------------------------------------------------------------------
template <int FLAGS>
__global__ __launch_bounds__(256)
void gemm_bf16_kernel(const float* __restrict__ A,
                      const __bf16* __restrict__ Wt,
                      const float* __restrict__ bias,
                      float* __restrict__ C,
                      int M, int K, int Nout) {
  __shared__ __bf16 sA[2][128][72];   // 64 + 8 pad, rows 144B (16B aligned)
  __shared__ __bf16 sB[2][64][72];

  const int tid  = threadIdx.x;
  const int wave = tid >> 5;
  const int lane = tid & 31;
  const int bm = blockIdx.y * 128;
  const int bn = blockIdx.x * 64;

  // staging coordinates
  const int ar  = tid >> 1;            // 0..127
  const int acg = (tid & 1) * 32;      // 0 / 32
  const int bnr = tid >> 2;            // 0..63
  const int bkg = (tid & 3) * 16;      // 0/16/32/48
  const float*  aptr = A  + (size_t)(bm + ar)  * K + acg;
  const __bf16* bptr = Wt + (size_t)(bn + bnr) * K + bkg;

  float4 areg[8];
  uint4  breg[2];
  auto gload = [&](int k0) {
    #pragma unroll
    for (int j = 0; j < 8; ++j) areg[j] = *(const float4*)(aptr + k0 + j * 4);
    breg[0] = *(const uint4*)(bptr + k0);
    breg[1] = *(const uint4*)(bptr + k0 + 8);
    __builtin_prefetch(aptr + k0 + 64, 0, 1);
  };
  auto lstore = [&](int buf) {
    #pragma unroll
    for (int j = 0; j < 8; ++j) {
      float4 v = areg[j];
      if (FLAGS & GF_RELU_IN) {
        v.x = fmaxf(v.x, 0.f); v.y = fmaxf(v.y, 0.f);
        v.z = fmaxf(v.z, 0.f); v.w = fmaxf(v.w, 0.f);
      }
      __bf16* d = &sA[buf][ar][acg + j * 4];
      d[0] = (__bf16)v.x; d[1] = (__bf16)v.y;
      d[2] = (__bf16)v.z; d[3] = (__bf16)v.w;
    }
    *(uint4*)&sB[buf][bnr][bkg]     = breg[0];
    *(uint4*)&sB[buf][bnr][bkg + 8] = breg[1];
  };

  v8f acc[4] = {v8f{}, v8f{}, v8f{}, v8f{}};
  const int m    = lane & 15;
  const int half = lane >> 4;

  gload(0);
  lstore(0);
  __syncthreads();

  const int nk = K / 64;
  int buf = 0;
  for (int i = 0; i < nk; ++i) {
    if (i + 1 < nk) gload((i + 1) * 64);   // overlap with compute below

    #pragma unroll
    for (int ks = 0; ks < 64; ks += 32) {
      v16bf afrag;
      {
        uint4* au = (uint4*)&afrag;
        au[0] = *(uint4*)&sA[buf][wave * 16 + m][ks + half * 16];
        au[1] = *(uint4*)&sA[buf][wave * 16 + m][ks + half * 16 + 8];
      }
      v16bf bfrag[4];
      #pragma unroll
      for (int nt = 0; nt < 4; ++nt) {
        uint4* bu = (uint4*)&bfrag[nt];
        bu[0] = *(uint4*)&sB[buf][nt * 16 + m][ks + half * 16];
        bu[1] = *(uint4*)&sB[buf][nt * 16 + m][ks + half * 16 + 8];
      }
      #pragma unroll
      for (int nt = 0; nt < 4; ++nt) {
        acc[nt] = __builtin_amdgcn_wmma_f32_16x16x32_bf16(
            false, afrag, false, bfrag[nt], (short)0, acc[nt], false, false);
      }
    }

    if (i + 1 < nk) {
      lstore(buf ^ 1);     // writes buffer last read in iter i-1 (fenced then)
      __syncthreads();
      buf ^= 1;
    }
  }

  // ---- epilogue ----
  const int nloc = lane & 15;
  const int mb   = (lane >> 4) * 8;
  #pragma unroll
  for (int nt = 0; nt < 4; ++nt) {
    int col = bn + nt * 16 + nloc;
    float bv = bias ? bias[col] : 0.f;
    #pragma unroll
    for (int r = 0; r < 8; ++r) {
      int row = bm + wave * 16 + mb + r;
      size_t off = (size_t)row * Nout + col;
      float v = acc[nt][r] + bv;
      if (FLAGS & GF_ACC)      v += C[off];
      if (FLAGS & GF_RELU_OUT) v = fmaxf(v, 0.f);
      C[off] = v;
    }
  }
}

// ---------------------------------------------------------------------------
// Fused MessagePassingUnit_v1.
//  x = cat(U,P) [2048]; LN -> relu -> (bf16) @ Wt[64][2048] -> +b -> sigmoid
//  gate = mean over 64; msg = P * gate.
// MODE 0: out[e] = 0.5*msg ; MODE 1: out[e] += 0.5*msg
// MODE 2: atomicAdd(out[tgt_idx[e]], msg); atomicAdd(cnt[tgt_idx[e]], 1)
// Block = 128 threads (4 waves) -> 16 edges; each wave owns one 16x16
// subtile of the [16 x 64] gate GEMM. BK=64, double-buffered LN staging.
// ---------------------------------------------------------------------------
template <int MODE>
__global__ __launch_bounds__(128)
void mpu_kernel(const float* __restrict__ uptr, const int* __restrict__ uidx,
                const float* __restrict__ pptr, const int* __restrict__ pidx,
                const float* __restrict__ ln_g, const float* __restrict__ ln_b,
                const __bf16* __restrict__ Wt,    // [64][2048] bf16
                const float* __restrict__ bmpu,   // [64]
                float* __restrict__ outp,
                float* __restrict__ cnt,
                const int* __restrict__ tgt_idx) {
  const int K2 = 2 * HH;
  __shared__ float  s_sum[16][8];
  __shared__ float  s_sq[16][8];
  __shared__ float  s_mu[16], s_rs[16], s_gate[16];
  __shared__ __bf16 sX[2][16][72];

  const int tid  = threadIdx.x;
  const int wave = tid >> 5;
  const int lane = tid & 31;
  const int e0   = blockIdx.x * 16;
  const int el   = tid >> 3;        // edge-local 0..15
  const int l8   = tid & 7;
  const int e    = e0 + el;

  const float* urow = uptr + (size_t)(uidx ? uidx[e] : e) * HH;
  const float* prow = pptr + (size_t)(pidx ? pidx[e] : e) * HH;

  // -------- phase 1: LN statistics (each of 8 threads covers 256 elems)
  {
    float s = 0.f, q = 0.f;
    int kbase = l8 * 256;
    const float* src = (kbase < HH) ? (urow + kbase) : (prow + kbase - HH);
    for (int j = 0; j < 256; j += 4) {
      float4 v = *(const float4*)(src + j);
      s += v.x + v.y + v.z + v.w;
      q += v.x * v.x + v.y * v.y + v.z * v.z + v.w * v.w;
    }
    s_sum[el][l8] = s; s_sq[el][l8] = q;
  }
  if (tid < 16) s_gate[tid] = 0.f;
  __syncthreads();
  if (tid < 16) {
    float ss = 0.f, qq = 0.f;
    #pragma unroll
    for (int j = 0; j < 8; ++j) { ss += s_sum[tid][j]; qq += s_sq[tid][j]; }
    float mu  = ss * (1.f / 2048.f);
    float var = qq * (1.f / 2048.f) - mu * mu;
    s_mu[tid] = mu;
    s_rs[tid] = rsqrtf(var + 1e-5f);
  }
  __syncthreads();

  // -------- phase 2: gate GEMM [16,2048] x [2048,64] via WMMA, BK=64
  const int m    = lane & 15;
  const int half = lane >> 4;
  const int kk   = l8 * 8;

  float4 xv[2];
  auto gloadx = [&](int k0) {
    int k = k0 + kk;
    const float* src = (k < HH) ? (urow + k) : (prow + (k - HH));
    xv[0] = *(const float4*)src;
    xv[1] = *(const float4*)(src + 4);
  };
  auto lstorex = [&](int buf, int k0) {
    int k = k0 + kk;
    float mu = s_mu[el], rs = s_rs[el];
    #pragma unroll
    for (int h = 0; h < 2; ++h) {
      float4 v = xv[h];
      int kj = k + h * 4;
      float x0 = fmaxf((v.x - mu) * rs * ln_g[kj + 0] + ln_b[kj + 0], 0.f);
      float x1 = fmaxf((v.y - mu) * rs * ln_g[kj + 1] + ln_b[kj + 1], 0.f);
      float x2 = fmaxf((v.z - mu) * rs * ln_g[kj + 2] + ln_b[kj + 2], 0.f);
      float x3 = fmaxf((v.w - mu) * rs * ln_g[kj + 3] + ln_b[kj + 3], 0.f);
      __bf16* d = &sX[buf][el][kk + h * 4];
      d[0] = (__bf16)x0; d[1] = (__bf16)x1;
      d[2] = (__bf16)x2; d[3] = (__bf16)x3;
    }
  };

  v8f acc = v8f{};
  gloadx(0);
  lstorex(0, 0);
  __syncthreads();

  const int nk = K2 / 64;
  int buf = 0;
  for (int i = 0; i < nk; ++i) {
    const int k0 = i * 64;
    if (i + 1 < nk) gloadx(k0 + 64);

    #pragma unroll
    for (int ks = 0; ks < 64; ks += 32) {
      v16bf a, b;
      uint4* au = (uint4*)&a;
      au[0] = *(uint4*)&sX[buf][m][ks + half * 16];
      au[1] = *(uint4*)&sX[buf][m][ks + half * 16 + 8];
      const __bf16* bsrc = Wt + (size_t)(wave * 16 + m) * K2 + k0 + ks + half * 16;
      uint4* bu = (uint4*)&b;
      bu[0] = *(const uint4*)bsrc;
      bu[1] = *(const uint4*)(bsrc + 8);
      acc = __builtin_amdgcn_wmma_f32_16x16x32_bf16(
          false, a, false, b, (short)0, acc, false, false);
    }

    if (i + 1 < nk) {
      lstorex(buf ^ 1, k0 + 64);
      __syncthreads();
      buf ^= 1;
    }
  }

  // -------- phase 3: sigmoid + mean over 64 columns -> gate[e]
  {
    int n  = wave * 16 + (lane & 15);
    int mb = (lane >> 4) * 8;
    float bv = bmpu[n];
    #pragma unroll
    for (int r = 0; r < 8; ++r) {
      float v = 1.f / (1.f + __expf(-(acc[r] + bv)));
      atomicAdd(&s_gate[mb + r], v);
    }
  }
  __syncthreads();
  const float gate = s_gate[el] * (1.f / 64.f);

  // -------- phase 4: msg = P * gate, written per MODE
  if (MODE == 2 && l8 == 0) atomicAdd(&cnt[tgt_idx[e]], 1.f);
  {
    int hbase = l8 * 128;
    const float* pr = prow + hbase;
    float* orow = (MODE == 2)
        ? outp + (size_t)tgt_idx[e] * HH + hbase
        : outp + (size_t)e * HH + hbase;
    for (int j = 0; j < 128; j += 4) {
      float4 pv = *(const float4*)(pr + j);
      if (MODE == 0) {
        float4 o = {0.5f * pv.x * gate, 0.5f * pv.y * gate,
                    0.5f * pv.z * gate, 0.5f * pv.w * gate};
        *(float4*)(orow + j) = o;
      } else if (MODE == 1) {
        float4 o = *(float4*)(orow + j);
        o.x += 0.5f * pv.x * gate; o.y += 0.5f * pv.y * gate;
        o.z += 0.5f * pv.z * gate; o.w += 0.5f * pv.w * gate;
        *(float4*)(orow + j) = o;
      } else {
        atomicAdd(&orow[j + 0], pv.x * gate);
        atomicAdd(&orow[j + 1], pv.y * gate);
        atomicAdd(&orow[j + 2], pv.z * gate);
        atomicAdd(&orow[j + 3], pv.w * gate);
      }
    }
  }
}

// obj_msg = 0.5*(sum1/max(cnt1,1) + sum2/max(cnt2,1))
__global__ void finalize_obj_msg_kernel(const float* __restrict__ sum1,
                                        const float* __restrict__ cnt1,
                                        const float* __restrict__ sum2,
                                        const float* __restrict__ cnt2,
                                        float* __restrict__ obj_msg) {
  long long i = (long long)blockIdx.x * 256 + threadIdx.x;
  if (i >= (long long)NN * HH) return;
  int n = (int)(i / HH);
  float c1 = fmaxf(cnt1[n], 1.f);
  float c2 = fmaxf(cnt2[n], 1.f);
  obj_msg[i] = 0.5f * (sum1[i] / c1 + sum2[i] / c2);
}

// ---------------------------------------------------------------------------
extern "C" void kernel_launch(void* const* d_in, const int* in_sizes, int n_in,
                              void* d_out, int out_size, void* d_ws, size_t ws_size,
                              hipStream_t stream) {
  const float* obj_feat   = (const float*)d_in[0];
  const float* rel_feat   = (const float*)d_in[1];
  const int*   sub_idx    = (const int*)d_in[2];
  const int*   obj_idx    = (const int*)d_in[3];
  const float* w_obj_down = (const float*)d_in[4];
  const float* b_obj_down = (const float*)d_in[5];
  const float* w_rel_down = (const float*)d_in[6];
  const float* b_rel_down = (const float*)d_in[7];
  const float* ln_g[4], *ln_b[4], *w_mpu[4], *b_mpu[4];
  for (int i = 0; i < 4; ++i) {           // s2p, o2p, p2s, p2o
    ln_g[i]  = (const float*)d_in[8 + 4 * i + 0];
    ln_b[i]  = (const float*)d_in[8 + 4 * i + 1];
    w_mpu[i] = (const float*)d_in[8 + 4 * i + 2];
    b_mpu[i] = (const float*)d_in[8 + 4 * i + 3];
  }
  const float* wih_objf = (const float*)d_in[24];
  const float* bih_objf = (const float*)d_in[25];
  const float* whh_objf = (const float*)d_in[26];
  const float* bhh_objf = (const float*)d_in[27];
  const float* wih_relf = (const float*)d_in[28];
  const float* bih_relf = (const float*)d_in[29];
  const float* whh_relf = (const float*)d_in[30];
  const float* bhh_relf = (const float*)d_in[31];

  // ---- workspace carve-out ----
  char* ws = (char*)d_ws;
  size_t off = 0;
  auto carve = [&](size_t bytes) -> void* {
    void* p = ws + off;
    off += (bytes + 255) & ~(size_t)255;
    return p;
  };
  float* obj_a   = (float*)carve((size_t)NN * HH * 4);
  float* obj_b   = (float*)carve((size_t)NN * HH * 4);
  float* rel_a   = (float*)carve((size_t)EE * HH * 4);
  float* rel_b   = (float*)carve((size_t)EE * HH * 4);
  float* rel_msg = (float*)carve((size_t)EE * HH * 4);
  float* obj_msg = (float*)carve((size_t)NN * HH * 4);
  float* sum1    = (float*)carve((size_t)NN * HH * 4);
  float* sum2    = (float*)carve((size_t)NN * HH * 4);
  float* cnt1    = (float*)carve((size_t)NN * 4);
  float* cnt2    = (float*)carve((size_t)NN * 4);
  __bf16* wt_obj_down = (__bf16*)carve((size_t)HH * PP * 2);
  __bf16* wt_rel_down = (__bf16*)carve((size_t)HH * PP * 2);
  __bf16* wt_mpu[4];
  for (int i = 0; i < 4; ++i) wt_mpu[i] = (__bf16*)carve((size_t)FF * 2 * HH * 2);
  __bf16* wt_wih_objf = (__bf16*)carve((size_t)HH * HH * 2);
  __bf16* wt_whh_objf = (__bf16*)carve((size_t)HH * HH * 2);
  __bf16* wt_wih_relf = (__bf16*)carve((size_t)HH * HH * 2);
  __bf16* wt_whh_relf = (__bf16*)carve((size_t)HH * HH * 2);

  auto tcvt = [&](const float* src, __bf16* dst, int K, int Nv) {
    long long tot = (long long)K * Nv;
    transpose_cvt_kernel<<<(int)((tot + 255) / 256), 256, 0, stream>>>(src, dst, K, Nv);
  };
  tcvt(w_obj_down, wt_obj_down, PP, HH);
  tcvt(w_rel_down, wt_rel_down, PP, HH);
  for (int i = 0; i < 4; ++i) tcvt(w_mpu[i], wt_mpu[i], 2 * HH, FF);
  tcvt(wih_objf, wt_wih_objf, HH, HH);
  tcvt(whh_objf, wt_whh_objf, HH, HH);
  tcvt(wih_relf, wt_wih_relf, HH, HH);
  tcvt(whh_relf, wt_whh_relf, HH, HH);

  auto gemm = [&](const float* A, const __bf16* Wt, const float* bias, float* C,
                  int M, int K, int Nout, int flags) {
    dim3 grid(Nout / 64, M / 128);
    switch (flags) {
      case GF_RELU_OUT:
        gemm_bf16_kernel<GF_RELU_OUT><<<grid, 256, 0, stream>>>(A, Wt, bias, C, M, K, Nout);
        break;
      case GF_RELU_IN:
        gemm_bf16_kernel<GF_RELU_IN><<<grid, 256, 0, stream>>>(A, Wt, bias, C, M, K, Nout);
        break;
      case GF_RELU_IN | GF_ACC:
        gemm_bf16_kernel<GF_RELU_IN | GF_ACC><<<grid, 256, 0, stream>>>(A, Wt, bias, C, M, K, Nout);
        break;
      default:
        gemm_bf16_kernel<0><<<grid, 256, 0, stream>>>(A, Wt, bias, C, M, K, Nout);
        break;
    }
  };
  auto zero = [&](float* p, long long n) {
    zero_f32_kernel<<<(int)((n + 255) / 256), 256, 0, stream>>>(p, n);
  };

  // down-dim projections with ReLU
  gemm(obj_feat, wt_obj_down, b_obj_down, obj_a, NN, PP, HH, GF_RELU_OUT);
  gemm(rel_feat, wt_rel_down, b_rel_down, rel_a, EE, PP, HH, GF_RELU_OUT);

  float* obj_cur = obj_a;
  float* rel_cur = rel_a;
  float* out_obj = (float*)d_out;
  float* out_rel = (float*)d_out + (size_t)NN * HH;

  for (int it = 0; it < 2; ++it) {
    zero(sum1, (long long)NN * HH);
    zero(sum2, (long long)NN * HH);
    zero(cnt1, NN);
    zero(cnt2, NN);

    const int mpu_blocks = EE / 16;
    // s2p: unary=rel, pair=obj[sub]    -> rel_msg  = 0.5*msg
    mpu_kernel<0><<<mpu_blocks, 128, 0, stream>>>(
        rel_cur, nullptr, obj_cur, sub_idx, ln_g[0], ln_b[0],
        wt_mpu[0], b_mpu[0], rel_msg, nullptr, nullptr);
    // o2p: unary=rel, pair=obj[obj]    -> rel_msg += 0.5*msg
    mpu_kernel<1><<<mpu_blocks, 128, 0, stream>>>(
        rel_cur, nullptr, obj_cur, obj_idx, ln_g[1], ln_b[1],
        wt_mpu[1], b_mpu[1], rel_msg, nullptr, nullptr);
    // p2s: unary=obj[sub], pair=rel    -> scatter-add into sum1 by sub_idx
    mpu_kernel<2><<<mpu_blocks, 128, 0, stream>>>(
        obj_cur, sub_idx, rel_cur, nullptr, ln_g[2], ln_b[2],
        wt_mpu[2], b_mpu[2], sum1, cnt1, sub_idx);
    // p2o: unary=obj[obj], pair=rel    -> scatter-add into sum2 by obj_idx
    mpu_kernel<2><<<mpu_blocks, 128, 0, stream>>>(
        obj_cur, obj_idx, rel_cur, nullptr, ln_g[3], ln_b[3],
        wt_mpu[3], b_mpu[3], sum2, cnt2, obj_idx);

    finalize_obj_msg_kernel<<<(int)(((long long)NN * HH + 255) / 256), 256, 0, stream>>>(
        sum1, cnt1, sum2, cnt2, obj_msg);

    float* obj_dst = (it == 1) ? out_obj : obj_b;
    float* rel_dst = (it == 1) ? out_rel : rel_b;

    // MessageFusion: relu(msg)@wih + bih + relu(state)@whh + bhh
    gemm(obj_msg, wt_wih_objf, bih_objf, obj_dst, NN, HH, HH, GF_RELU_IN);
    gemm(obj_cur, wt_whh_objf, bhh_objf, obj_dst, NN, HH, HH, GF_RELU_IN | GF_ACC);
    gemm(rel_msg, wt_wih_relf, bih_relf, rel_dst, EE, HH, HH, GF_RELU_IN);
    gemm(rel_cur, wt_whh_relf, bhh_relf, rel_dst, EE, HH, HH, GF_RELU_IN | GF_ACC);

    obj_cur = obj_dst;
    rel_cur = rel_dst;
  }
}
